// MIPNetwork_39582418600071
// MI455X (gfx1250) — compile-verified, matrix operand order checked by reading, not run on hardware
//
#include <hip/hip_runtime.h>
#include <hip/hip_bf16.h>
#include <math.h>

// ---------------------------------------------------------------------------
// Types for CDNA5 WMMA
// ---------------------------------------------------------------------------
typedef __attribute__((ext_vector_type(16))) __bf16 v16bf;
typedef __attribute__((ext_vector_type(8)))  float  v8f;

#define WAVE 32

static __host__ __device__ inline int ceil_div(int a, int b) { return (a + b - 1) / b; }

// ---------------------------------------------------------------------------
// Pack a row-major fp32 weight matrix W[K][N] into WMMA B-fragment order:
//   frag index = (kb*NT + nt)*32 + lane, each frag = 16 bf16 (32 bytes).
// Lane packing follows the CDNA5 16-bit B(32x16) VGPR layout; K-tail zeros
// baked in so the GEMM hot loop needs no bounds checks.
// ---------------------------------------------------------------------------
__global__ void pack_w_kernel(const float* __restrict__ W, int K, int N,
                              __bf16* __restrict__ out) {
    int NT = N >> 4;
    int KB = (K + 31) >> 5;
    int total = KB * NT * 32;
    int idx = blockIdx.x * blockDim.x + threadIdx.x;
    if (idx >= total) return;
    int lane = idx & 31;
    int t    = idx >> 5;
    int nt   = t % NT;
    int kb   = t / NT;
    int g  = lane >> 4, lm = lane & 15;
    int n  = nt * 16 + lm;
    __bf16* o = out + (size_t)idx * 16;
#pragma unroll
    for (int r = 0; r < 8; ++r) {
        int k = (kb << 5) + ((r < 4) ? (8 * g + 2 * r) : (16 + 8 * g + 2 * (r - 4)));
        o[2 * r]     = (k     < K) ? (__bf16)W[(size_t)k * N + n]       : (__bf16)0.f;
        o[2 * r + 1] = (k + 1 < K) ? (__bf16)W[(size_t)(k + 1) * N + n] : (__bf16)0.f;
    }
}

// ---------------------------------------------------------------------------
// variables[row][f] = relaxed[row]
// ---------------------------------------------------------------------------
__global__ void bcast_init_kernel(const float* __restrict__ relaxed, float* __restrict__ vars, int total) {
    int i = blockIdx.x * blockDim.x + threadIdx.x;
    if (i < total) vars[i] = relaxed[i >> 6];
}

// out[idx[e]] += val[e]^2
__global__ void seg_sq_sum_kernel(const float* __restrict__ val, const int* __restrict__ idx,
                                  int nnz, float* __restrict__ out) {
    int e = blockIdx.x * blockDim.x + threadIdx.x;
    if (e < nnz) atomicAdd(&out[idx[e]], val[e] * val[e]);
}

__global__ void sqrt_eps_kernel(float* __restrict__ x, int n) {
    int i = blockIdx.x * blockDim.x + threadIdx.x;
    if (i < n) x[i] = sqrtf(x[i] + 1e-6f);
}

// acc += sum(x^2)   (objective normalization)
__global__ void sumsq_vec_kernel(const float* __restrict__ x, int n, float* __restrict__ acc) {
    int i = blockIdx.x * blockDim.x + threadIdx.x;
    if (i < n) atomicAdd(acc, x[i] * x[i]);
}

__global__ void normalize_obj_kernel(const float* __restrict__ obj, int n,
                                     const float* __restrict__ sumsq, float* __restrict__ out) {
    int i = blockIdx.x * blockDim.x + threadIdx.x;
    if (i < n) {
        float denom = sqrtf(sumsq[0] / (float)n) + 1e-6f;
        out[i] = obj[i] / denom;
    }
}

// ---------------------------------------------------------------------------
// Generic 64-feature segment gather/scatter (one wave per edge):
//   out[sidx[e]*outStride + outOff + f] += val[e] * feat[gidx[e]*featStride + featOff + f]
// ---------------------------------------------------------------------------
__global__ void seg_scatter64_kernel(const float* __restrict__ feat, int featStride, int featOff,
                                     const int* __restrict__ gidx, const int* __restrict__ sidx,
                                     const float* __restrict__ val, int nnz,
                                     float* __restrict__ out, int outStride, int outOff) {
    int e = blockIdx.x * (blockDim.x >> 5) + (threadIdx.x >> 5);
    if (e >= nnz) return;
    int lane = threadIdx.x & 31;
    int s = gidx[e], d = sidx[e];
    float w = val[e];
    const float* fs = feat + (size_t)s * featStride + featOff;
    float* od = out + (size_t)d * outStride + outOff;
    atomicAdd(&od[lane],      w * fs[lane]);
    atomicAdd(&od[lane + 32], w * fs[lane + 32]);
}

// msg_pos/msg_neg fused scatter into cat_vu (stride 258, offsets 64 / 128)
__global__ void seg_msg_posneg_kernel(const float* __restrict__ ctmp,
                                      const int* __restrict__ ineq_dst, const int* __restrict__ ineq_src,
                                      const float* __restrict__ val, int nnz,
                                      float* __restrict__ catvu) {
    int e = blockIdx.x * (blockDim.x >> 5) + (threadIdx.x >> 5);
    if (e >= nnz) return;
    int lane = threadIdx.x & 31;
    int d = ineq_dst[e], s = ineq_src[e];
    float w  = val[e];
    float vp = fmaxf(w, 0.f);
    float vn = fmaxf(-w, 0.f);
    const float* cf = ctmp + (size_t)d * 128 + 64;
    float* o = catvu + (size_t)s * 258;
    float c0 = cf[lane], c1 = cf[lane + 32];
    atomicAdd(&o[64 + lane],       vp * c0);
    atomicAdd(&o[64 + lane + 32],  vp * c1);
    atomicAdd(&o[128 + lane],      vn * c0);
    atomicAdd(&o[128 + lane + 32], vn * c1);
}

// cat_q[row, 0:64] = variables ; cat_q[row, 64:68] = noise_i[row, 0:4]
__global__ void build_cat_q_kernel(const float* __restrict__ vars, const float* __restrict__ noise_i,
                                   float* __restrict__ cat, int total /* n*68 */) {
    int i = blockIdx.x * blockDim.x + threadIdx.x;
    if (i >= total) return;
    int row = i / 68, col = i - row * 68;
    cat[i] = (col < 64) ? vars[(size_t)row * 64 + col] : noise_i[(size_t)row * 4 + (col - 64)];
}

// cat_cu[row, 0:64]=constraints; cat_cu[row, 64:128]=(lhs - cv)/const_scaler
__global__ void finish_cat_cu_kernel(float* __restrict__ cat, const float* __restrict__ cons,
                                     const float* __restrict__ cv, const float* __restrict__ cscaler,
                                     int total /* n*64 */) {
    int i = blockIdx.x * blockDim.x + threadIdx.x;
    if (i >= total) return;
    int row = i >> 6, col = i & 63;
    size_t base = (size_t)row * 128;
    cat[base + col] = cons[(size_t)row * 64 + col];
    float x = cat[base + 64 + col];
    cat[base + 64 + col] = (x - cv[row]) / cscaler[row];
}

// cat_ec[row, 64:128] = ((ecv - eq_lhs)/eq_scaler)^2
__global__ void finish_cat_ec_kernel(float* __restrict__ cat, const float* __restrict__ eqc,
                                     const float* __restrict__ ecv, const float* __restrict__ escaler,
                                     int total /* n*64 */) {
    int i = blockIdx.x * blockDim.x + threadIdx.x;
    if (i >= total) return;
    int row = i >> 6, col = i & 63;
    size_t base = (size_t)row * 128;
    cat[base + col] = eqc[(size_t)row * 64 + col];
    float t = (ecv[row] - cat[base + 64 + col]) / escaler[row];
    cat[base + 64 + col] = t * t;
}

// cat_vu[row, 0:64] = variables ; cols 256/257 = obj_n / int_mask
__global__ void seed_cat_vu_kernel(const float* __restrict__ vars, const float* __restrict__ obj_n,
                                   const float* __restrict__ imask, float* __restrict__ cat,
                                   int total /* n*64 */) {
    int i = blockIdx.x * blockDim.x + threadIdx.x;
    if (i >= total) return;
    int row = i >> 6, col = i & 63;
    size_t base = (size_t)row * 258;
    cat[base + col] = vars[(size_t)row * 64 + col];
    if (col == 0) {
        cat[base + 256] = obj_n[row];
        cat[base + 257] = imask[row];
    }
}

// cat_vu[row, 64:192] /= vars_scaler[row]
__global__ void scale_msgs_kernel(float* __restrict__ cat, const float* __restrict__ vscaler,
                                  int total /* n*128 */) {
    int i = blockIdx.x * blockDim.x + threadIdx.x;
    if (i >= total) return;
    int row = i >> 7, col = i & 127;
    cat[(size_t)row * 258 + 64 + col] /= vscaler[row];
}

// out[row,0:64] = tmp[row, 0:64](ldt) + 0.5 * out[row,0:64]
__global__ void residual_update_kernel(float* __restrict__ out, const float* __restrict__ tmp,
                                       int ldt, int total /* n*64 */) {
    int i = blockIdx.x * blockDim.x + threadIdx.x;
    if (i >= total) return;
    int row = i >> 6, col = i & 63;
    out[i] = tmp[(size_t)row * ldt + col] + 0.5f * out[i];
}

// Pairnorm finish: scale = rsqrt(1e-6 + sumsq/nrows); reset accumulator.
__global__ void pairnorm_finish_kernel(float* __restrict__ s, float nrowsInv) {
    s[1] = rsqrtf(1e-6f + s[0] * nrowsInv);
    s[0] = 0.f;
}

// ---------------------------------------------------------------------------
// WMMA GEMM: Y[M, NT*16] = op(A[M,K]) @ Wpacked + bias   (bf16 x bf16 -> f32)
// One wave per 16-row strip; A fragment loaded/converted once per K-block and
// reused across NT column tiles. Full K-blocks use guard-free aligned float2
// loads (all lda here are even); only the final partial block (K=68/258) takes
// the guarded path. B fragments are contiguous 32-byte v16bf loads.
//   PROLOG=1: op(a) = relu(a * (*scalePtr))   (fused pairnorm scale + relu)
//   EPILOG=1: Y += 0.5 * extra[row]
// ---------------------------------------------------------------------------
template <int NT, int PROLOG, int EPILOG>
__global__ void wmma_gemm_kernel(const float* __restrict__ A, int lda,
                                 const v16bf* __restrict__ Wp,
                                 const float* __restrict__ bias,
                                 float* __restrict__ Y, int ldy,
                                 int M, int K,
                                 const float* __restrict__ scalePtr,
                                 const float* __restrict__ extra,
                                 float* __restrict__ sumsqAcc) {
    const int lane  = threadIdx.x & 31;
    const int g     = lane >> 4;
    const int lm    = lane & 15;
    const int mBase = blockIdx.x << 4;
    int arow = mBase + lm;
    if (arow >= M) arow = M - 1;          // clamp: loads stay in-bounds, no branch

    float scale = 1.0f;
    if (PROLOG) scale = scalePtr[0];

    v8f acc[NT];
#pragma unroll
    for (int nt = 0; nt < NT; ++nt) {
        float bv = bias[nt * 16 + lm];
#pragma unroll
        for (int r = 0; r < 8; ++r) acc[nt][r] = bv;
    }

    const int KFULL = K >> 5;
    const int KB    = (K + 31) >> 5;
    const float* Arow = A + (size_t)arow * lda;

    for (int kb = 0; kb < KFULL; ++kb) {
        const int k0 = kb << 5;
        // ---- A fragment: guard-free aligned float2 loads ----
        const float2* p = (const float2*)(Arow + k0 + 8 * g);
        v16bf a;
#pragma unroll
        for (int r = 0; r < 8; ++r) {
            float2 x = (r < 4) ? p[r] : p[8 + (r - 4)];
            if (PROLOG) { x.x = fmaxf(x.x * scale, 0.f); x.y = fmaxf(x.y * scale, 0.f); }
            a[2 * r]     = (__bf16)x.x;
            a[2 * r + 1] = (__bf16)x.y;
        }
        const v16bf* bp = Wp + (size_t)kb * NT * 32 + lane;
#pragma unroll
        for (int nt = 0; nt < NT; ++nt) {
            v16bf b = bp[nt * 32];
            acc[nt] = __builtin_amdgcn_wmma_f32_16x16x32_bf16(false, a, false, b,
                                                              (short)0, acc[nt], false, false);
        }
    }

    if (KB > KFULL) {                      // single guarded tail block
        const int kb = KFULL, k0 = kb << 5;
        v16bf a;
#pragma unroll
        for (int r = 0; r < 8; ++r) {
            int k = k0 + ((r < 4) ? (8 * g + 2 * r) : (16 + 8 * g + 2 * (r - 4)));
            float x0 = (k     < K) ? Arow[k]     : 0.f;
            float x1 = (k + 1 < K) ? Arow[k + 1] : 0.f;
            if (PROLOG) { x0 = fmaxf(x0 * scale, 0.f); x1 = fmaxf(x1 * scale, 0.f); }
            a[2 * r]     = (__bf16)x0;
            a[2 * r + 1] = (__bf16)x1;
        }
        const v16bf* bp = Wp + (size_t)kb * NT * 32 + lane;
#pragma unroll
        for (int nt = 0; nt < NT; ++nt) {
            v16bf b = bp[nt * 32];
            acc[nt] = __builtin_amdgcn_wmma_f32_16x16x32_bf16(false, a, false, b,
                                                              (short)0, acc[nt], false, false);
        }
    }

    float lsum = 0.f;
#pragma unroll
    for (int nt = 0; nt < NT; ++nt) {
#pragma unroll
        for (int r = 0; r < 8; ++r) {
            int row = mBase + r + 8 * g;   // C/D layout
            if (row < M) {
                float v = acc[nt][r];
                if (EPILOG) v += 0.5f * extra[row];
                Y[(size_t)row * ldy + nt * 16 + lm] = v;
                lsum += v * v;
            }
        }
    }
    if (sumsqAcc) atomicAdd(sumsqAcc, lsum);
}

// ---------------------------------------------------------------------------
// Output head layer 2 + sigmoid mixing
// ---------------------------------------------------------------------------
__global__ void o_final_kernel(const float* __restrict__ H, const float* __restrict__ scalePtr,
                               const float* __restrict__ o_w2, const float* __restrict__ o_b2,
                               const float* __restrict__ imask, float* __restrict__ d_out,
                               int n, int pass, int steps) {
    int row = blockIdx.x * blockDim.x + threadIdx.x;
    if (row >= n) return;
    float s = scalePtr[0];
    float acc = o_b2[0];
    const float* h = H + (size_t)row * 64;
#pragma unroll
    for (int k = 0; k < 64; ++k) acc += fmaxf(h[k] * s, 0.f) * o_w2[k];
    float m   = imask[row];
    float sig = 1.f / (1.f + expf(-acc));
    d_out[(size_t)pass * n + row]  = sig * m + acc * (1.f - m) * 0.1f;
    d_out[(size_t)steps * n + row] = acc;   // out_vars (last pass survives)
}

// ---------------------------------------------------------------------------
// Host orchestration
// ---------------------------------------------------------------------------
extern "C" void kernel_launch(void* const* d_in, const int* in_sizes, int n_in,
                              void* d_out_v, int out_size, void* d_ws, size_t ws_size,
                              hipStream_t stream) {
    (void)n_in; (void)out_size; (void)ws_size;

    const int*   ineq_src = (const int*)  d_in[0];
    const int*   ineq_dst = (const int*)  d_in[1];
    const float* ineq_val = (const float*)d_in[2];
    const int*   eq_src   = (const int*)  d_in[3];
    const int*   eq_dst   = (const int*)  d_in[4];
    const float* eq_val   = (const float*)d_in[5];
    const float* cv       = (const float*)d_in[6];
    const float* ecv      = (const float*)d_in[7];
    const float* relaxed  = (const float*)d_in[8];
    const float* obj      = (const float*)d_in[9];
    const float* imask    = (const float*)d_in[10];
    const float* noise    = (const float*)d_in[11];
    const float* q_w1  = (const float*)d_in[12]; const float* q_b1  = (const float*)d_in[13];
    const float* q_w2  = (const float*)d_in[14]; const float* q_b2  = (const float*)d_in[15];
    const float* cu_w1 = (const float*)d_in[16]; const float* cu_b1 = (const float*)d_in[17];
    const float* cu_w2 = (const float*)d_in[18]; const float* cu_b2 = (const float*)d_in[19];
    const float* ec_w1 = (const float*)d_in[20]; const float* ec_b1 = (const float*)d_in[21];
    const float* ec_w2 = (const float*)d_in[22]; const float* ec_b2 = (const float*)d_in[23];
    const float* vu_w1 = (const float*)d_in[24]; const float* vu_b1 = (const float*)d_in[25];
    const float* vu_w2 = (const float*)d_in[26]; const float* vu_b2 = (const float*)d_in[27];
    const float* o_w1  = (const float*)d_in[28]; const float* o_b1  = (const float*)d_in[29];
    const float* o_w2  = (const float*)d_in[30]; const float* o_b2  = (const float*)d_in[31];

    float* d_out = (float*)d_out_v;

    const int NNZ    = in_sizes[0];
    const int EQNNZ  = in_sizes[3];
    const int CONN   = in_sizes[6];
    const int EQN    = in_sizes[7];
    const int VARN   = in_sizes[8];
    const int STEPS  = in_sizes[11] / (VARN * 4);

    // ---- workspace carve (fp32 region, then 32B-aligned bf16 pack region) ----
    float* base = (float*)d_ws;
    size_t off = 0;
    auto F32 = [&](size_t n) { float* r = base + off; off += n; return r; };
    float* variables = F32((size_t)VARN * 64);
    float* cons      = F32((size_t)CONN * 64);
    float* eqcons    = F32((size_t)EQN  * 64);
    float* query     = F32((size_t)VARN * 64);   // reused as vu output
    float* Hbuf      = F32((size_t)VARN * 64);   // hidden buffer (max rows)
    float* ctmp      = F32((size_t)CONN * 128);
    float* etmp      = F32((size_t)EQN  * 128);
    float* cat_q     = F32((size_t)VARN * 68);
    float* cat_cu    = F32((size_t)CONN * 128);
    float* cat_ec    = F32((size_t)EQN  * 128);
    float* cat_vu    = F32((size_t)VARN * 258);
    float* cscaler   = F32(CONN);
    float* escaler   = F32(EQN);
    float* vscaler   = F32(VARN);
    float* obj_n     = F32(VARN);
    float* scal      = F32(8);                   // [0]=sumsq acc, [1]=scale, [2]=obj sumsq

    off = (off + 7) & ~(size_t)7;                // 32-byte align bf16 pack region
    __bf16* bbase = (__bf16*)(base + off);
    size_t boff = 0;
    auto packElems = [](int K, int N) { return (size_t)((K + 31) / 32) * (N / 16) * 32 * 16; };
    auto B16 = [&](int K, int N) { __bf16* r = bbase + boff; boff += packElems(K, N); return r; };
    __bf16* pqW1  = B16(68, 64);
    __bf16* pqW2  = B16(64, 64);
    __bf16* pcuW1 = B16(128, 64);
    __bf16* pcuW2 = B16(64, 128);
    __bf16* pecW1 = B16(128, 64);
    __bf16* pecW2 = B16(64, 128);
    __bf16* pvuW1 = B16(258, 64);
    __bf16* pvuW2 = B16(64, 64);
    __bf16* poW1  = B16(64, 64);

    const int TB = 256;
    auto g1 = [&](int n) { return dim3(ceil_div(n, TB)); };
    auto gw = [&](int nnz) { return dim3(ceil_div(nnz, TB / 32)); };  // wave-per-edge kernels
    auto gp = [&](int K, int N) { return dim3(ceil_div((int)(((K + 31) / 32) * (N / 16) * 32), TB)); };

    // ---- one-time (per launch) prep ----
    hipMemsetAsync(cons,    0, (size_t)CONN * 64 * sizeof(float), stream);
    hipMemsetAsync(eqcons,  0, (size_t)EQN  * 64 * sizeof(float), stream);
    hipMemsetAsync(cscaler, 0, CONN * sizeof(float), stream);
    hipMemsetAsync(escaler, 0, EQN  * sizeof(float), stream);
    hipMemsetAsync(vscaler, 0, VARN * sizeof(float), stream);
    hipMemsetAsync(scal,    0, 8 * sizeof(float), stream);

    pack_w_kernel<<<gp(68, 64),  TB, 0, stream>>>(q_w1,  68, 64,  pqW1);
    pack_w_kernel<<<gp(64, 64),  TB, 0, stream>>>(q_w2,  64, 64,  pqW2);
    pack_w_kernel<<<gp(128, 64), TB, 0, stream>>>(cu_w1, 128, 64, pcuW1);
    pack_w_kernel<<<gp(64, 128), TB, 0, stream>>>(cu_w2, 64, 128, pcuW2);
    pack_w_kernel<<<gp(128, 64), TB, 0, stream>>>(ec_w1, 128, 64, pecW1);
    pack_w_kernel<<<gp(64, 128), TB, 0, stream>>>(ec_w2, 64, 128, pecW2);
    pack_w_kernel<<<gp(258, 64), TB, 0, stream>>>(vu_w1, 258, 64, pvuW1);
    pack_w_kernel<<<gp(64, 64),  TB, 0, stream>>>(vu_w2, 64, 64,  pvuW2);
    pack_w_kernel<<<gp(64, 64),  TB, 0, stream>>>(o_w1,  64, 64,  poW1);

    // objective normalization
    sumsq_vec_kernel<<<g1(VARN), TB, 0, stream>>>(obj, VARN, scal + 2);
    normalize_obj_kernel<<<g1(VARN), TB, 0, stream>>>(obj, VARN, scal + 2, obj_n);

    // variables broadcast; initial constraint embeddings; scalers
    bcast_init_kernel<<<g1(VARN * 64), TB, 0, stream>>>(relaxed, variables, VARN * 64);
    seg_scatter64_kernel<<<gw(NNZ), TB, 0, stream>>>(variables, 64, 0, ineq_src, ineq_dst,
                                                     ineq_val, NNZ, cons, 64, 0);
    seg_scatter64_kernel<<<gw(EQNNZ), TB, 0, stream>>>(variables, 64, 0, eq_src, eq_dst,
                                                       eq_val, EQNNZ, eqcons, 64, 0);
    seg_sq_sum_kernel<<<g1(NNZ),   TB, 0, stream>>>(ineq_val, ineq_dst, NNZ,   cscaler);
    seg_sq_sum_kernel<<<g1(EQNNZ), TB, 0, stream>>>(eq_val,   eq_dst,   EQNNZ, escaler);
    seg_sq_sum_kernel<<<g1(NNZ),   TB, 0, stream>>>(ineq_val, ineq_src, NNZ,   vscaler);
    sqrt_eps_kernel<<<g1(CONN), TB, 0, stream>>>(cscaler, CONN);
    sqrt_eps_kernel<<<g1(EQN),  TB, 0, stream>>>(escaler, EQN);
    sqrt_eps_kernel<<<g1(VARN), TB, 0, stream>>>(vscaler, VARN);

    dim3 gv(ceil_div(VARN, 16));
    dim3 gc(ceil_div(CONN, 16));
    dim3 ge(ceil_div(EQN, 16));

    for (int ip = 0; ip < STEPS; ++ip) {
        const float* noise_i = noise + (size_t)ip * VARN * 4;

        // ---- query MLP (68 -> 64 -> 64), epilogue += 0.5*int_mask ----
        build_cat_q_kernel<<<g1(VARN * 68), TB, 0, stream>>>(variables, noise_i, cat_q, VARN * 68);
        wmma_gemm_kernel<4, 0, 0><<<gv, WAVE, 0, stream>>>(cat_q, 68, (const v16bf*)pqW1, q_b1,
                                                           Hbuf, 64, VARN, 68, nullptr, nullptr, scal);
        pairnorm_finish_kernel<<<1, 1, 0, stream>>>(scal, 1.f / (float)VARN);
        wmma_gemm_kernel<4, 1, 1><<<gv, WAVE, 0, stream>>>(Hbuf, 64, (const v16bf*)pqW2, q_b2,
                                                           query, 64, VARN, 64, scal + 1, imask, nullptr);

        // ---- ineq constraint update (128 -> 64 -> 128) ----
        hipMemsetAsync(cat_cu, 0, (size_t)CONN * 128 * sizeof(float), stream);
        seg_scatter64_kernel<<<gw(NNZ), TB, 0, stream>>>(query, 64, 0, ineq_src, ineq_dst,
                                                         ineq_val, NNZ, cat_cu, 128, 64);
        finish_cat_cu_kernel<<<g1(CONN * 64), TB, 0, stream>>>(cat_cu, cons, cv, cscaler, CONN * 64);
        wmma_gemm_kernel<4, 0, 0><<<gc, WAVE, 0, stream>>>(cat_cu, 128, (const v16bf*)pcuW1, cu_b1,
                                                           Hbuf, 64, CONN, 128, nullptr, nullptr, scal);
        pairnorm_finish_kernel<<<1, 1, 0, stream>>>(scal, 1.f / (float)CONN);
        wmma_gemm_kernel<8, 1, 0><<<gc, WAVE, 0, stream>>>(Hbuf, 64, (const v16bf*)pcuW2, cu_b2,
                                                           ctmp, 128, CONN, 64, scal + 1, nullptr, nullptr);
        residual_update_kernel<<<g1(CONN * 64), TB, 0, stream>>>(cons, ctmp, 128, CONN * 64);

        // ---- variable message buffer (cat_vu), fused msg scatters ----
        hipMemsetAsync(cat_vu, 0, (size_t)VARN * 258 * sizeof(float), stream);
        seed_cat_vu_kernel<<<g1(VARN * 64), TB, 0, stream>>>(variables, obj_n, imask, cat_vu, VARN * 64);
        seg_msg_posneg_kernel<<<gw(NNZ), TB, 0, stream>>>(ctmp, ineq_dst, ineq_src,
                                                          ineq_val, NNZ, cat_vu);
        scale_msgs_kernel<<<g1(VARN * 128), TB, 0, stream>>>(cat_vu, vscaler, VARN * 128);

        // ---- eq constraint update (128 -> 64 -> 128) ----
        hipMemsetAsync(cat_ec, 0, (size_t)EQN * 128 * sizeof(float), stream);
        seg_scatter64_kernel<<<gw(EQNNZ), TB, 0, stream>>>(query, 64, 0, eq_src, eq_dst,
                                                           eq_val, EQNNZ, cat_ec, 128, 64);
        finish_cat_ec_kernel<<<g1(EQN * 64), TB, 0, stream>>>(cat_ec, eqcons, ecv, escaler, EQN * 64);
        wmma_gemm_kernel<4, 0, 0><<<ge, WAVE, 0, stream>>>(cat_ec, 128, (const v16bf*)pecW1, ec_b1,
                                                           Hbuf, 64, EQN, 128, nullptr, nullptr, scal);
        pairnorm_finish_kernel<<<1, 1, 0, stream>>>(scal, 1.f / (float)EQN);
        wmma_gemm_kernel<8, 1, 0><<<ge, WAVE, 0, stream>>>(Hbuf, 64, (const v16bf*)pecW2, ec_b2,
                                                           etmp, 128, EQN, 64, scal + 1, nullptr, nullptr);
        residual_update_kernel<<<g1(EQN * 64), TB, 0, stream>>>(eqcons, etmp, 128, EQN * 64);

        // eq2var into cat_vu cols 192:256
        seg_scatter64_kernel<<<gw(EQNNZ), TB, 0, stream>>>(etmp, 128, 64, eq_dst, eq_src,
                                                           eq_val, EQNNZ, cat_vu, 258, 192);

        // ---- variable update MLP (258 -> 64 -> 64), residual ----
        wmma_gemm_kernel<4, 0, 0><<<gv, WAVE, 0, stream>>>(cat_vu, 258, (const v16bf*)pvuW1, vu_b1,
                                                           Hbuf, 64, VARN, 258, nullptr, nullptr, scal);
        pairnorm_finish_kernel<<<1, 1, 0, stream>>>(scal, 1.f / (float)VARN);
        wmma_gemm_kernel<4, 1, 0><<<gv, WAVE, 0, stream>>>(Hbuf, 64, (const v16bf*)pvuW2, vu_b2,
                                                           query, 64, VARN, 64, scal + 1, nullptr, nullptr);
        residual_update_kernel<<<g1(VARN * 64), TB, 0, stream>>>(variables, query, 64, VARN * 64);

        // ---- output head (64 -> 64 -> 1) + sigmoid mix ----
        wmma_gemm_kernel<4, 0, 0><<<gv, WAVE, 0, stream>>>(variables, 64, (const v16bf*)poW1, o_b1,
                                                           Hbuf, 64, VARN, 64, nullptr, nullptr, scal);
        pairnorm_finish_kernel<<<1, 1, 0, stream>>>(scal, 1.f / (float)VARN);
        o_final_kernel<<<g1(VARN), TB, 0, stream>>>(Hbuf, scal + 1, o_w2, o_b2, imask,
                                                    d_out, VARN, ip, STEPS);
    }
}
